// Net_24850680775442
// MI455X (gfx1250) — matrix operand order repaired and checked by reading, not compiled, not run on hardware
//
#include <hip/hip_runtime.h>
#include <math.h>

typedef _Float16 v16h __attribute__((ext_vector_type(16)));
typedef _Float16 v8h  __attribute__((ext_vector_type(8)));
typedef _Float16 v2h  __attribute__((ext_vector_type(2)));
typedef float    v8f  __attribute__((ext_vector_type(8)));

#define N_NODES 8192
#define BMW     256      // bitmap words per row = 8192/32
#define KP1     2048     // nodes after pool 0
#define KP2     1024     // nodes after pool 1

// ---------------------------------------------------------------- utilities
__global__ void k_zero_u32(unsigned* __restrict__ p, int n) {
    int t = blockIdx.x * blockDim.x + threadIdx.x;
    if (t < n) p[t] = 0u;
}

// A0[dst,src] = 1 (duplicates collapse, matching .at[].set(1.0))
__global__ void k_build_bitmap(const int* __restrict__ ei, int E, unsigned* __restrict__ bm) {
    int e = blockIdx.x * blockDim.x + threadIdx.x;
    if (e >= E) return;
    int src = ei[e], dst = ei[E + e];
    atomicOr(&bm[(size_t)dst * BMW + (src >> 5)], 1u << (src & 31));
}

__global__ void k_selfloop(unsigned* __restrict__ bm) {
    int i = blockIdx.x * blockDim.x + threadIdx.x;
    if (i < N_NODES) bm[(size_t)i * BMW + (i >> 5)] |= 1u << (i & 31);
}

// stores RECIPROCAL of clamped degree (multiply later instead of divide)
__global__ void k_deg_pop(const unsigned* __restrict__ bm, float* __restrict__ rdeg) {
    int i = blockIdx.x * blockDim.x + threadIdx.x;
    if (i >= N_NODES) return;
    int s = 0;
    const unsigned* r = bm + (size_t)i * BMW;
    for (int w = 0; w < BMW; ++w) s += __popc(r[w]);
    rdeg[i] = 1.0f / ((s > 1) ? (float)s : 1.0f);
}

// out[row,f] = sum_{j: A0[row,j]=1} x[j,f]   (optionally * rdeg[row])
__global__ void k_agg_bitmap(const unsigned* __restrict__ bm, const float* __restrict__ x,
                             const float* __restrict__ rdeg, float* __restrict__ out, int F) {
    int row = blockIdx.y;
    int f   = blockIdx.x * blockDim.x + threadIdx.x;
    const unsigned* r = bm + (size_t)row * BMW;
    float acc = 0.f;
    for (int w = 0; w < BMW; ++w) {
        unsigned u = r[w];
        while (u) {
            int b = __ffs(u) - 1;
            u &= u - 1u;
            acc += x[(size_t)(w * 32 + b) * F + f];
        }
    }
    out[(size_t)row * F + f] = rdeg ? (acc * rdeg[row]) : acc;
}

// -------------------------------------------------------------- WMMA GEMM
// 8-wave workgroup computes a 64x64 tile of C = A*B (+bias, *rrowdiv, act).
// A staged in LDS as 64x32 f16 [m][k]; B staged TRANSPOSED as 64x32 f16 [n][k]
// so both fragments are two contiguous 16B ds_load_b128 reads per lane
// (CDNA5 ISA 7.12.2 lane layouts, wave32). Each wave: 1 M-tile x 2 N-tiles,
// sharing one A fragment across two v_wmma_f32_16x16x32_f16 per k-step.
template <typename TA, typename TB>
__global__ __launch_bounds__(256)
void k_gemm_wmma(const TA* __restrict__ A, const TB* __restrict__ B, float* __restrict__ C,
                 int M, int Nc, int K, int lda, int ldb, int ldc,
                 const float* __restrict__ bias, const float* __restrict__ rrowdiv,
                 int accload, int act) {
    (void)M; (void)Nc;
    __shared__ _Float16 sA[64 * 32];   // [m][k]
    __shared__ _Float16 sB[64 * 32];   // [n][k] (transposed B)

    const int tid  = threadIdx.x;
    const int wave = tid >> 5;
    const int lane = tid & 31;
    const int lm   = lane & 15;
    const int hi   = lane >> 4;
    const int tm   = wave & 3;            // M sub-tile 0..3
    const int tn0  = (wave >> 2) * 2;     // N sub-tile pair base: 0 or 2
    const int m0   = blockIdx.y * 64;
    const int n0   = blockIdx.x * 64;

    v8f acc0 = {}, acc1 = {};
    if (accload) {
#pragma unroll
        for (int v = 0; v < 8; ++v) {
            const int m = m0 + tm * 16 + v + hi * 8;
            acc0[v] = C[(size_t)m * ldc + n0 + tn0 * 16 + lm];
            acc1[v] = C[(size_t)m * ldc + n0 + (tn0 + 1) * 16 + lm];
        }
    }

    for (int k0 = 0; k0 < K; k0 += 32) {
        __syncthreads();
        // stage A: 64x32, f32->f16 on the fly; each thread 2 rows x 4 cols
        // (packed ds_store_b64 per 4 halves)
        {
            const int r  = tid >> 3;          // 0..31
            const int cc = (tid & 7) * 4;     // 0,4,..,28
#pragma unroll
            for (int rr = 0; rr < 2; ++rr) {
                const int row = r + rr * 32;
                const TA* src = A + (size_t)(m0 + row) * lda + k0 + cc;
                _Float16* dst = sA + row * 32 + cc;
                dst[0] = (_Float16)src[0]; dst[1] = (_Float16)src[1];
                dst[2] = (_Float16)src[2]; dst[3] = (_Float16)src[3];
            }
        }
        // stage B transposed: B[k][n] -> sB[n*32+k].
        // Each thread owns a k-pair (adjacent in sB) x 4 n-cols: the two halves
        // pack into one 32-bit LDS store (4x ds_store_b32 instead of 8x b16).
        {
            const int kr = (tid >> 4) * 2;    // 0,2,..,30
            const int cn = (tid & 15) * 4;    // 0,4,..,60
            const TB* s0 = B + (size_t)(k0 + kr) * ldb + n0 + cn;
            const TB* s1 = s0 + ldb;
#pragma unroll
            for (int q = 0; q < 4; ++q) {
                v2h pr;
                pr[0] = (_Float16)s0[q];
                pr[1] = (_Float16)s1[q];
                *(v2h*)(sB + (cn + q) * 32 + kr) = pr;
            }
        }
        // prefetch next k-panel into cache (lowers to global_prefetch_b8)
        if (k0 + 32 < K) {
            __builtin_prefetch(A + (size_t)(m0 + (tid >> 3)) * lda + k0 + 32, 0, 1);
            __builtin_prefetch(B + (size_t)(k0 + 32 + (tid >> 3)) * ldb + n0, 0, 1);
        }
        __syncthreads();

        // A fragment: lane lm = row; elements 0-7 at k=hi*8, 8-15 at k=16+hi*8
        const _Float16* ra = sA + (tm * 16 + lm) * 32;
        v8h alo = *(const v8h*)(ra + hi * 8);
        v8h ahi = *(const v8h*)(ra + 16 + hi * 8);
        v16h a = __builtin_shufflevector(alo, ahi, 0, 1, 2, 3, 4, 5, 6, 7,
                                         8, 9, 10, 11, 12, 13, 14, 15);
        // B fragments: lane lm = col n; elements 0-7 at k=hi*8, 8-15 at k=16+hi*8
        const _Float16* rb0 = sB + (tn0 * 16 + lm) * 32 + hi * 8;
        v16h b0 = __builtin_shufflevector(*(const v8h*)rb0, *(const v8h*)(rb0 + 16),
                                          0, 1, 2, 3, 4, 5, 6, 7,
                                          8, 9, 10, 11, 12, 13, 14, 15);
        const _Float16* rb1 = sB + ((tn0 + 1) * 16 + lm) * 32 + hi * 8;
        v16h b1 = __builtin_shufflevector(*(const v8h*)rb1, *(const v8h*)(rb1 + 16),
                                          0, 1, 2, 3, 4, 5, 6, 7,
                                          8, 9, 10, 11, 12, 13, 14, 15);

        acc0 = __builtin_amdgcn_wmma_f32_16x16x32_f16(false, a, false, b0, (short)0, acc0, false, false);
        acc1 = __builtin_amdgcn_wmma_f32_16x16x32_f16(false, a, false, b1, (short)0, acc1, false, false);
    }

#pragma unroll
    for (int t = 0; t < 2; ++t) {
        const v8f cc = t ? acc1 : acc0;
        const int n = n0 + (tn0 + t) * 16 + lm;
#pragma unroll
        for (int v = 0; v < 8; ++v) {
            const int m = m0 + tm * 16 + v + hi * 8;
            float r = cc[v];
            if (rrowdiv) r *= rrowdiv[m];
            if (bias)    r += bias[n];
            if (act == 1)      r = (r > 0.f) ? r : 0.01f * r;  // leaky_relu
            else if (act == 2) r = (r > 0.f) ? r : 0.f;        // relu
            C[(size_t)m * ldc + n] = r;
        }
    }
}

// GraphConv scorer: score[i] = <aggraw[i], wrel> + brel + <x[i], wroot>
__global__ void k_score(const float* __restrict__ aggraw, const float* __restrict__ x,
                        const float* __restrict__ wrel, const float* __restrict__ brel,
                        const float* __restrict__ wroot, float* __restrict__ score,
                        int n, int F) {
    int node = blockIdx.x * (blockDim.x >> 5) + (threadIdx.x >> 5);
    int lane = threadIdx.x & 31;
    if (node >= n) return;
    float acc = 0.f;
    for (int c = lane; c < F; c += 32)
        acc += aggraw[(size_t)node * F + c] * wrel[c] + x[(size_t)node * F + c] * wroot[c];
    for (int o = 16; o; o >>= 1) acc += __shfl_xor(acc, o, 32);
    if (lane == 0) score[node] = acc + brel[0];
}

// Full descending bitonic sort of (score, idx) in LDS; stable tiebreak on idx.
__global__ void k_bitonic(const float* __restrict__ score, int n,
                          float* __restrict__ keys, int* __restrict__ idxs) {
    extern __shared__ char smem[];
    float* sk = (float*)smem;
    int*   si = (int*)(smem + (size_t)n * sizeof(float));
    for (int i = threadIdx.x; i < n; i += blockDim.x) { sk[i] = score[i]; si[i] = i; }
    __syncthreads();
    for (int size = 2; size <= n; size <<= 1) {
        for (int stride = size >> 1; stride > 0; stride >>= 1) {
            for (int i = threadIdx.x; i < n; i += blockDim.x) {
                int j = i ^ stride;
                if (j > i) {
                    bool desc = ((i & size) == 0);
                    float ki = sk[i], kj = sk[j];
                    int   ii = si[i], ij = si[j];
                    bool inorder = (ki > kj) || (ki == kj && ii < ij);
                    if (inorder != desc) { sk[i] = kj; sk[j] = ki; si[i] = ij; si[j] = ii; }
                }
            }
            __syncthreads();
        }
    }
    for (int i = threadIdx.x; i < n; i += blockDim.x) { keys[i] = sk[i]; idxs[i] = si[i]; }
}

// out[i,:] = f[perm[i],:] * tanh(sv[i])
__global__ void k_pool(const float* __restrict__ f, int F, const int* __restrict__ perm,
                       const float* __restrict__ keys, float* __restrict__ out, int k) {
    int t = blockIdx.x * blockDim.x + threadIdx.x;
    if (t >= k * F) return;
    int i = t / F, c = t - i * F;
    out[t] = f[(size_t)perm[i] * F + c] * tanhf(keys[i]);
}

// A1raw[i,j] = A0[perm0[i], perm0[j]]  as f16 {0,1}
__global__ void k_A1raw(const unsigned* __restrict__ bm, const int* __restrict__ perm,
                        _Float16* __restrict__ out) {
    int j = blockIdx.x * blockDim.x + threadIdx.x;
    int i = blockIdx.y;
    if (j >= KP1) return;
    int pi = perm[i], pj = perm[j];
    unsigned bit = (bm[(size_t)pi * BMW + (pj >> 5)] >> (pj & 31)) & 1u;
    out[(size_t)i * KP1 + j] = (_Float16)(float)bit;
}

__global__ void k_thresh(const float* __restrict__ M1, _Float16* __restrict__ A1h, int n) {
    int t = blockIdx.x * blockDim.x + threadIdx.x;
    if (t >= n) return;
    A1h[t] = (M1[t] > 0.f) ? (_Float16)1.0f : (_Float16)0.0f;
}

// stores RECIPROCAL of clamped row-sum
__global__ void k_rowsum_h(const _Float16* __restrict__ A, int k, float* __restrict__ rdeg, int rows) {
    int row  = blockIdx.x * (blockDim.x >> 5) + (threadIdx.x >> 5);
    int lane = threadIdx.x & 31;
    if (row >= rows) return;
    float s = 0.f;
    for (int j = lane; j < k; j += 32) s += (float)A[(size_t)row * k + j];
    for (int o = 16; o; o >>= 1) s += __shfl_xor(s, o, 32);
    if (lane == 0) rdeg[row] = 1.0f / ((s > 1.f) ? s : 1.f);
}

// dst[perm[i],:] = src[i,:]   (unpool scatter)
__global__ void k_scatter(const float* __restrict__ src, const int* __restrict__ perm,
                          float* __restrict__ dst, int k, int F) {
    int t = blockIdx.x * blockDim.x + threadIdx.x;
    if (t >= k * F) return;
    int i = t / F, c = t - i * F;
    dst[(size_t)perm[i] * F + c] = src[t];
}

// A2[i,j] = A1[perm1[i], perm1[j]]  -> float output
__global__ void k_A2(const _Float16* __restrict__ A1h, const int* __restrict__ perm,
                     float* __restrict__ out) {
    int j = blockIdx.x * blockDim.x + threadIdx.x;
    int i = blockIdx.y;
    if (j >= KP2) return;
    out[(size_t)i * KP2 + j] = (float)A1h[(size_t)perm[i] * KP1 + perm[j]];
}

// ---------------------------------------------------------------- driver
extern "C" void kernel_launch(void* const* d_in, const int* in_sizes, int n_in,
                              void* d_out, int out_size, void* d_ws, size_t ws_size,
                              hipStream_t stream) {
    (void)n_in; (void)out_size; (void)ws_size;
    const float* x     = (const float*)d_in[0];
    const float* W0l   = (const float*)d_in[1];
    const float* b0    = (const float*)d_in[2];
    const float* W0r   = (const float*)d_in[3];
    const float* W1l   = (const float*)d_in[4];
    const float* b1    = (const float*)d_in[5];
    const float* W1r   = (const float*)d_in[6];
    const float* U0l   = (const float*)d_in[7];
    const float* c0    = (const float*)d_in[8];
    const float* U0r   = (const float*)d_in[9];
    const float* U1l   = (const float*)d_in[10];
    const float* c1    = (const float*)d_in[11];
    const float* U1r   = (const float*)d_in[12];
    const float* P0rel = (const float*)d_in[13];
    const float* P0b   = (const float*)d_in[14];
    const float* P0rt  = (const float*)d_in[15];
    const float* P1rel = (const float*)d_in[16];
    const float* P1b   = (const float*)d_in[17];
    const float* P1rt  = (const float*)d_in[18];
    const int*   ei    = (const int*)d_in[19];
    const int E = in_sizes[19] / 2;

    float* out_z      = (float*)d_out;                       // 8192 x 128
    float* out_latent = out_z + (size_t)N_NODES * 128;       // 1024 x 128
    float* out_adj    = out_latent + (size_t)KP2 * 128;      // 1024 x 1024
    float* out_b      = out_adj + (size_t)KP2 * KP2;         // 1024 (int32 zeros)

    // ---- workspace carve ----
    char* p = (char*)d_ws;
    auto carve = [&](size_t bytes) -> char* {
        char* r = p; p += (bytes + 255) & ~(size_t)255; return r;
    };
    unsigned*  bm0   = (unsigned*)carve((size_t)N_NODES * BMW * 4);   // 8 MB adjacency bitmap
    float*     rdeg0 = (float*)   carve((size_t)N_NODES * 4);
    float*     fA    = (float*)   carve((size_t)N_NODES * 256 * 4);   // SAGE0 output
    float*     aggA  = (float*)   carve((size_t)N_NODES * 256 * 4);   // A0 aggregation scratch
    float*     sc0   = (float*)   carve((size_t)N_NODES * 4);
    float*     skey0 = (float*)   carve((size_t)N_NODES * 4);
    int*       sidx0 = (int*)     carve((size_t)N_NODES * 4);
    float*     fp    = (float*)   carve((size_t)KP1 * 256 * 4);       // pooled features
    _Float16*  A1raw = (_Float16*)carve((size_t)KP1 * KP1 * 2);
    float*     M1    = (float*)   carve((size_t)KP1 * KP1 * 4);       // A1raw @ A1raw
    _Float16*  A1h   = (_Float16*)carve((size_t)KP1 * KP1 * 2);       // A1 (0/1) f16
    float*     rdeg1 = (float*)   carve((size_t)KP1 * 4);
    float*     agg1  = (float*)   carve((size_t)KP1 * 256 * 4);       // A1 aggregation scratch
    float*     f1    = (float*)   carve((size_t)KP1 * 128 * 4);
    float*     sc1   = (float*)   carve((size_t)KP1 * 4);
    float*     skey1 = (float*)   carve((size_t)KP1 * 4);
    int*       sidx1 = (int*)     carve((size_t)KP1 * 4);
    float*     up1   = (float*)   carve((size_t)KP1 * 128 * 4);
    float*     z1    = (float*)   carve((size_t)KP1 * 256 * 4);
    float*     up2   = (float*)   carve((size_t)N_NODES * 256 * 4);

    // ---- A0 bitmap + reciprocal degrees ----
    { int n = N_NODES * BMW; k_zero_u32<<<(n + 255) / 256, 256, 0, stream>>>(bm0, n); }
    k_build_bitmap<<<(E + 255) / 256, 256, 0, stream>>>(ei, E, bm0);
    k_selfloop<<<(N_NODES + 255) / 256, 256, 0, stream>>>(bm0);
    k_deg_pop<<<(N_NODES + 255) / 256, 256, 0, stream>>>(bm0, rdeg0);

    // ---- down SAGE0: f = leaky( mean(A0,x)@W0l + b0 + x@W0r ) ----
    k_agg_bitmap<<<dim3(1, N_NODES), 128, 0, stream>>>(bm0, x, rdeg0, aggA, 128);
    k_gemm_wmma<float, float><<<dim3(256 / 64, N_NODES / 64), 256, 0, stream>>>(
        aggA, W0l, fA, N_NODES, 256, 128, 128, 256, 256, nullptr, nullptr, 0, 0);
    k_gemm_wmma<float, float><<<dim3(256 / 64, N_NODES / 64), 256, 0, stream>>>(
        x, W0r, fA, N_NODES, 256, 128, 128, 256, 256, b0, nullptr, 1, 1);

    // ---- SAGPool 0 (sum-aggregation scorer, then top-K1) ----
    k_agg_bitmap<<<dim3(2, N_NODES), 128, 0, stream>>>(bm0, fA, nullptr, aggA, 256);
    k_score<<<N_NODES / 4, 128, 0, stream>>>(aggA, fA, P0rel, P0b, P0rt, sc0, N_NODES, 256);
    k_bitonic<<<1, 1024, (size_t)N_NODES * 8, stream>>>(sc0, N_NODES, skey0, sidx0);
    k_pool<<<(KP1 * 256 + 255) / 256, 256, 0, stream>>>(fA, 256, sidx0, skey0, fp, KP1);
    k_A1raw<<<dim3(KP1 / 256, KP1), 256, 0, stream>>>(bm0, sidx0, A1raw);

    // ---- A1 = (A1raw @ A1raw > 0): exact {0,1} f16 inputs, f32 accumulate ----
    k_gemm_wmma<_Float16, _Float16><<<dim3(KP1 / 64, KP1 / 64), 256, 0, stream>>>(
        A1raw, A1raw, M1, KP1, KP1, KP1, KP1, KP1, KP1, nullptr, nullptr, 0, 0);
    k_thresh<<<(KP1 * KP1 + 255) / 256, 256, 0, stream>>>(M1, A1h, KP1 * KP1);
    k_rowsum_h<<<KP1 / 4, 128, 0, stream>>>(A1h, KP1, rdeg1, KP1);

    // ---- down SAGE1 ----
    k_gemm_wmma<_Float16, float><<<dim3(256 / 64, KP1 / 64), 256, 0, stream>>>(
        A1h, fp, agg1, KP1, 256, KP1, KP1, 256, 256, nullptr, rdeg1, 0, 0);   // mean agg
    k_gemm_wmma<float, float><<<dim3(128 / 64, KP1 / 64), 256, 0, stream>>>(
        agg1, W1l, f1, KP1, 128, 256, 256, 128, 128, nullptr, nullptr, 0, 0);
    k_gemm_wmma<float, float><<<dim3(128 / 64, KP1 / 64), 256, 0, stream>>>(
        fp, W1r, f1, KP1, 128, 256, 256, 128, 128, b1, nullptr, 1, 1);

    // ---- SAGPool 1 ----
    k_gemm_wmma<_Float16, float><<<dim3(128 / 64, KP1 / 64), 256, 0, stream>>>(
        A1h, f1, agg1, KP1, 128, KP1, KP1, 128, 128, nullptr, nullptr, 0, 0); // sum agg
    k_score<<<KP1 / 4, 128, 0, stream>>>(agg1, f1, P1rel, P1b, P1rt, sc1, KP1, 128);
    k_bitonic<<<1, 1024, (size_t)KP1 * 8, stream>>>(sc1, KP1, skey1, sidx1);
    k_pool<<<(KP2 * 128 + 255) / 256, 256, 0, stream>>>(f1, 128, sidx1, skey1, out_latent, KP2);
    k_A2<<<dim3(KP2 / 256, KP2), 256, 0, stream>>>(A1h, sidx1, out_adj);
    k_zero_u32<<<(KP2 + 255) / 256, 256, 0, stream>>>((unsigned*)out_b, KP2);

    // ---- up SAGE0: unpool to K1, relu(_sage(up, A1, U0l, c0, U0r)) ----
    { int n = KP1 * 128; k_zero_u32<<<(n + 255) / 256, 256, 0, stream>>>((unsigned*)up1, n); }
    k_scatter<<<(KP2 * 128 + 255) / 256, 256, 0, stream>>>(out_latent, sidx1, up1, KP2, 128);
    k_gemm_wmma<_Float16, float><<<dim3(128 / 64, KP1 / 64), 256, 0, stream>>>(
        A1h, up1, agg1, KP1, 128, KP1, KP1, 128, 128, nullptr, rdeg1, 0, 0);  // mean agg
    k_gemm_wmma<float, float><<<dim3(256 / 64, KP1 / 64), 256, 0, stream>>>(
        agg1, U0l, z1, KP1, 256, 128, 128, 256, 256, nullptr, nullptr, 0, 0);
    k_gemm_wmma<float, float><<<dim3(256 / 64, KP1 / 64), 256, 0, stream>>>(
        up1, U0r, z1, KP1, 256, 128, 128, 256, 256, c0, nullptr, 1, 2);

    // ---- up SAGE1: unpool to N, _sage(up2, A0, U1l, c1, U1r) -> out_z ----
    { int n = N_NODES * 256; k_zero_u32<<<(n + 255) / 256, 256, 0, stream>>>((unsigned*)up2, n); }
    k_scatter<<<(KP1 * 256 + 255) / 256, 256, 0, stream>>>(z1, sidx0, up2, KP1, 256);
    k_agg_bitmap<<<dim3(2, N_NODES), 128, 0, stream>>>(bm0, up2, rdeg0, aggA, 256);
    k_gemm_wmma<float, float><<<dim3(128 / 64, N_NODES / 64), 256, 0, stream>>>(
        aggA, U1l, out_z, N_NODES, 128, 256, 256, 128, 128, nullptr, nullptr, 0, 0);
    k_gemm_wmma<float, float><<<dim3(128 / 64, N_NODES / 64), 256, 0, stream>>>(
        up2, U1r, out_z, N_NODES, 128, 256, 256, 128, 128, c1, nullptr, 1, 0);
}
